// ElectronGNN_22600117911704
// MI455X (gfx1250) — matrix operand order repaired and checked by reading, not compiled, not run on hardware
//
#include <hip/hip_runtime.h>
#include <hip/hip_bf16.h>
#include <math.h>

#define NB 512
#define NE 64
#define NN 16
#define FF 8
#define DD 128
#define NL 3

#define DDP (DD + 8)   // padded row length (u16) for row-major LDS tiles
#define NET (NE + 8)   // padded row length (u16) for transposed x
#define YNT 40         // padded row length (u16) for transposed y (K=32 + pad)

typedef __attribute__((ext_vector_type(16))) __bf16 v16bf;
typedef __attribute__((ext_vector_type(8)))  float  v8f;
typedef __attribute__((ext_vector_type(4)))  unsigned int u32x4;

union BF16Frag {
  v16bf v;
  unsigned short u[16];
  unsigned int w[8];
  u32x4 q[2];
};

__device__ __forceinline__ unsigned short f2bf(float x) {
  unsigned int b = __float_as_uint(x);
  b += 0x7FFFu + ((b >> 16) & 1u);
  return (unsigned short)(b >> 16);
}

__device__ __forceinline__ float bf2f(unsigned short u) {
  return __uint_as_float(((unsigned int)u) << 16);
}

// fast tanh: 1 - 2/(e^{2x}+1); saturates correctly at +-inf
__device__ __forceinline__ float fast_tanh(float x) {
  const float ez = __expf(2.0f * x);
  return 1.0f - 2.0f * __builtin_amdgcn_rcpf(ez + 1.0f);
}

// 16-bit A-matrix (16x32) per-lane K index for element p (0..15), hi = lane/16
__device__ __forceinline__ int a_kofp(int p, int hi) {
  int v = p >> 1, h = p & 1;
  return (v < 4) ? (8 * hi + 2 * v + h) : (16 + 8 * hi + 2 * (v - 4) + h);
}

__device__ __forceinline__ v8f wmma_bf16(v16bf a, v16bf b, v8f c) {
  return __builtin_amdgcn_wmma_f32_16x16x32_bf16(false, a, false, b, (short)0, c,
                                                 false, false);
}

__device__ __forceinline__ float rbf_dist(const float* pa, const float* pb) {
  const float dx = pa[0] - pb[0];
  const float dy = pa[1] - pb[1];
  const float dz = pa[2] - pb[2];
  return __builtin_amdgcn_sqrtf(dx * dx + dy * dy + dz * dz + 1e-12f);
}

// ---------------- w_up bf16 fragment pre-pack ----------------
// fragment order: (((l*4 + part)*4 + kt)*8 + nt)*32 + lane, 16 bf16 each.
#define WPK_ELEMS (3 * 4 * 4 * 8 * 32 * 16)
#define WPK_BYTES ((size_t)WPK_ELEMS * 2)

__global__ __launch_bounds__(256) void pack_wup_kernel(
    const float* __restrict__ w_up, unsigned short* __restrict__ wpk) {
  const int gid = blockIdx.x * 256 + threadIdx.x;  // one thread = (frag, lane)
  if (gid >= 3 * 4 * 4 * 8 * 32) return;
  const int lane = gid & 31;
  const int frag = gid >> 5;
  const int nt = frag & 7;
  const int kt = (frag >> 3) & 3;
  const int part = (frag >> 5) & 3;
  const int l = frag >> 7;
  const int hi = lane >> 4, nl = lane & 15;
  const int dcol = nt * 16 + nl;
  BF16Frag f;
#pragma unroll
  for (int p = 0; p < 16; ++p) {
    const int krow = part * DD + kt * 32 + a_kofp(p, hi);
    f.u[p] = f2bf(w_up[((size_t)l * 4 * DD + krow) * DD + dcol]);
  }
  u32x4* dst = (u32x4*)(wpk + (size_t)gid * 16);
  dst[0] = f.q[0];
  dst[1] = f.q[1];
}

// ---------------- message GEMM phase ----------------
// mode 0 = same-spin ee, 1 = anti-spin ee, 2 = ne
// z[i,d] = sum_f ( sum_k A[(i,f),k] * B[k,d] ) * we[f,d]  -> zbuf (bf16)
__device__ __forceinline__ void message_phase(
    int mode, const float* __restrict__ we, unsigned short (*xbfT)[NET],
    unsigned short (*zbuf)[DDP], unsigned short (*ybfT)[YNT], float (*rl)[3],
    float (*Rl)[3], int w, int lane) {
  const int hi = lane >> 4;
  const int nl = lane & 15;
  const int mlo = lane & 15;          // row within 16-row A tile
  const int fi = mlo & 7;             // f of this lane's A rows
  const float mu = (4.0f / 7.0f) * (float)fi;
  const int nkt = (mode == 2) ? 1 : 2;

#pragma unroll 1
  for (int q = 0; q < 4; ++q) {
    const int mt = w * 4 + q;               // 32 M-tiles over 8 waves
    const int iA = 2 * mt + (mlo >> 3);     // electron index for A rows
    const v8f zero = {0, 0, 0, 0, 0, 0, 0, 0};
    v8f acc[8];
#pragma unroll
    for (int nt = 0; nt < 8; ++nt) acc[nt] = zero;

#pragma unroll 1
    for (int kt = 0; kt < nkt; ++kt) {
      BF16Frag A;
#pragma unroll
      for (int p = 0; p < 16; ++p) {
        const int k = kt * 32 + a_kofp(p, hi);
        float val = 0.0f;
        if (mode == 2) {
          if (k < NN) {
            const float dist = rbf_dist(&rl[iA][0], &Rl[k][0]);
            const float e = dist - mu;
            val = __expf(-e * e);
          }
        } else {
          const int j = k;
          const bool ss = ((iA >= 32) == (j >= 32));
          const bool valid = (j != iA) && ((mode == 0) ? ss : !ss);
          if (valid) {
            const float dist = rbf_dist(&rl[iA][0], &rl[j][0]);
            const float e = dist - mu;
            val = __expf(-e * e);
          }
        }
        A.u[p] = f2bf(val);
      }
      const int k0 = kt * 32 + 8 * hi;
#pragma unroll
      for (int nt = 0; nt < 8; ++nt) {
        const int dcol = nt * 16 + nl;
        // B element p <-> k = k0 + p (p<8), k0 + 16 + (p-8) (p>=8) -> two 16B LDS loads
        const unsigned short* brow =
            (mode == 2) ? &ybfT[dcol][0] : &xbfT[dcol][0];
        BF16Frag Bf;
        Bf.q[0] = *(const u32x4*)(brow + k0);
        Bf.q[1] = *(const u32x4*)(brow + k0 + 16);
        acc[nt] = wmma_bf16(A.v, Bf.v, acc[nt]);
      }
    }
    // C layout: VGPR v <-> M = v + 8*hi = (i_local*8 + f)  => f = v, i_local = hi
    const int iZ = 2 * mt + hi;
#pragma unroll
    for (int nt = 0; nt < 8; ++nt) {
      const int dcol = nt * 16 + nl;
      float z = 0.0f;
#pragma unroll
      for (int v = 0; v < 8; ++v) z += acc[nt][v] * we[v * DD + dcol];
      zbuf[iZ][dcol] = f2bf(z);
    }
  }
}

// ---------------- one K=128 slab of the update GEMM ----------------
// accU += A_part[64,128] @ w_up[part*128:+128, :]
// pre-packed-weights variant: branch-free hot loop
__device__ __forceinline__ void update_part_pk(
    int part, v8f* accU, const unsigned short* __restrict__ wpk_l,
    const unsigned short* arow, int ntU0, int lane) {
  const int hi = lane >> 4;
#pragma unroll 1
  for (int kt = 0; kt < 4; ++kt) {
    BF16Frag A;
    // A pairs are contiguous: k = 8*hi .. +7 and 16+8*hi .. +7 -> two 16B loads
    A.q[0] = *(const u32x4*)(arow + kt * 32 + 8 * hi);
    A.q[1] = *(const u32x4*)(arow + kt * 32 + 16 + 8 * hi);
#pragma unroll
    for (int t = 0; t < 4; ++t) {
      const unsigned short* src =
          wpk_l + ((size_t)(((part * 4 + kt) * 8 + (ntU0 + t))) * 32 + lane) * 16;
      const u32x4* s4 = (const u32x4*)src;
      BF16Frag Bf;
      Bf.q[0] = s4[0];
      Bf.q[1] = s4[1];
      accU[t] = wmma_bf16(A.v, Bf.v, accU[t]);
    }
  }
}

// direct-from-global fallback (ws too small)
__device__ __forceinline__ void update_part_direct(
    int part, v8f* accU, const float* __restrict__ wup_l,
    const unsigned short* arow, int ntU0, int lane) {
  const int hi = lane >> 4;
  const int nl = lane & 15;
#pragma unroll 1
  for (int kt = 0; kt < 4; ++kt) {
    BF16Frag A;
    A.q[0] = *(const u32x4*)(arow + kt * 32 + 8 * hi);
    A.q[1] = *(const u32x4*)(arow + kt * 32 + 16 + 8 * hi);
#pragma unroll
    for (int t = 0; t < 4; ++t) {
      const int dcol = (ntU0 + t) * 16 + nl;
      BF16Frag Bf;
#pragma unroll
      for (int p = 0; p < 16; ++p) {
        const int krow = part * DD + kt * 32 + a_kofp(p, hi);
        Bf.u[p] = f2bf(wup_l[krow * DD + dcol]);
      }
      accU[t] = wmma_bf16(A.v, Bf.v, accU[t]);
    }
  }
}

__device__ __forceinline__ void update_part(int part, v8f* accU, int useWs,
                                            const unsigned short* __restrict__ wpk_l,
                                            const float* __restrict__ wup_l,
                                            const unsigned short* arow, int ntU0,
                                            int lane) {
  if (useWs) {
    update_part_pk(part, accU, wpk_l, arow, ntU0, lane);
  } else {
    update_part_direct(part, accU, wup_l, arow, ntU0, lane);
  }
}

__global__ __launch_bounds__(256) void egnn_kernel(
    const float* __restrict__ r, const float* __restrict__ R,
    const float* __restrict__ embed, const float* __restrict__ we_same,
    const float* __restrict__ we_anti, const float* __restrict__ we_ne,
    const float* __restrict__ w_up, const float* __restrict__ b_up,
    const int* __restrict__ atom_types, const unsigned short* __restrict__ wpk,
    int useWs, float* __restrict__ out) {
  __shared__ __align__(16) unsigned short xbf[NE][DDP];    // 17 KB  x, row-major (A operand)
  __shared__ __align__(16) unsigned short xbfT[DD][NET];   // 18 KB  x, transposed (B operand)
  __shared__ __align__(16) unsigned short zbuf[NE][DDP];   // 17 KB  messages, row-major
  __shared__ __align__(16) unsigned short ybfT[DD][YNT];   // 10 KB  y, transposed, K padded
  __shared__ float rl[NE][3];
  __shared__ float Rl[NN][3];

  const int tid = threadIdx.x;
  const int w = tid >> 5;
  const int lane = tid & 31;
  const int hi = lane >> 4;
  const int nl = lane & 15;
  const int b = blockIdx.x;

  for (int t = tid; t < NE * 3; t += 256) rl[t / 3][t % 3] = r[b * NE * 3 + t];
  for (int t = tid; t < NN * 3; t += 256) Rl[t / 3][t % 3] = R[b * NN * 3 + t];
  for (int t = tid; t < 32 * DD; t += 256) {
    const int n = t / DD, d = t % DD;
    const float v = (n < NN) ? embed[atom_types[n] * DD + d] : 0.0f;
    ybfT[d][n] = f2bf(v);
  }
  __syncthreads();

  const float MU = 4.0f / 7.0f;
  // x0 = ne_feat reshape: x[i, n*8+f] = exp(-(|r_i - R_n| - mu_f)^2)
  for (int t = tid; t < NE * DD; t += 256) {
    const int i = t / DD, dc = t % DD;
    const int n = dc >> 3, f = dc & 7;
    const float dist = rbf_dist(&rl[i][0], &Rl[n][0]);
    const float e = dist - MU * (float)f;
    const unsigned short xv = f2bf(__expf(-e * e));
    xbf[i][dc] = xv;
    xbfT[dc][i] = xv;
  }
  __syncthreads();

  const int mtU = w >> 1;          // update-GEMM: wave owns one 16-row M tile
  const int ntU0 = (w & 1) * 4;    // ... and four 16-col N tiles
  const int iU = mtU * 16 + nl;    // A row for the update GEMM

#pragma unroll 1
  for (int l = 0; l < NL; ++l) {
    const float* wup_l = w_up + (size_t)l * 4 * DD * DD;
    const unsigned short* wpk_l = wpk + (size_t)l * 4 * 4 * 8 * 32 * 16;
    const v8f zero = {0, 0, 0, 0, 0, 0, 0, 0};
    v8f accU[4];
#pragma unroll
    for (int t = 0; t < 4; ++t) accU[t] = zero;

    // part 0: x @ W[0:128]
    update_part(0, accU, useWs, wpk_l, wup_l, &xbf[iU][0], ntU0, lane);

    // z_same -> zbuf, then accumulate part 1
    message_phase(0, we_same + l * FF * DD, xbfT, zbuf, ybfT, rl, Rl, w, lane);
    __syncthreads();
    update_part(1, accU, useWs, wpk_l, wup_l, &zbuf[iU][0], ntU0, lane);
    __syncthreads();

    // z_anti -> zbuf, then accumulate part 2
    message_phase(1, we_anti + l * FF * DD, xbfT, zbuf, ybfT, rl, Rl, w, lane);
    __syncthreads();
    update_part(2, accU, useWs, wpk_l, wup_l, &zbuf[iU][0], ntU0, lane);
    __syncthreads();

    // z_ne -> zbuf, then accumulate part 3
    message_phase(2, we_ne + l * FF * DD, xbfT, zbuf, ybfT, rl, Rl, w, lane);
    __syncthreads();
    update_part(3, accU, useWs, wpk_l, wup_l, &zbuf[iU][0], ntU0, lane);
    __syncthreads();

    // residual: x += tanh(upd @ W + b)  (update both layouts of x)
    {
      const float* bl = b_up + l * DD;
#pragma unroll
      for (int t = 0; t < 4; ++t) {
        const int dcol = (ntU0 + t) * 16 + nl;
        const float bb = bl[dcol];
#pragma unroll
        for (int v = 0; v < 8; ++v) {
          const int i = mtU * 16 + v + 8 * hi;
          const float xv = bf2f(xbf[i][dcol]);
          const unsigned short nv = f2bf(xv + fast_tanh(accU[t][v] + bb));
          xbf[i][dcol] = nv;
          xbfT[dcol][i] = nv;
        }
      }
    }
    __syncthreads();
  }

  for (int t = tid; t < NE * DD; t += 256)
    out[(size_t)b * NE * DD + t] = bf2f(xbf[t / DD][t % DD]);
}

extern "C" void kernel_launch(void* const* d_in, const int* in_sizes, int n_in,
                              void* d_out, int out_size, void* d_ws,
                              size_t ws_size, hipStream_t stream) {
  (void)in_sizes; (void)n_in; (void)out_size;
  const float* r = (const float*)d_in[0];
  const float* R = (const float*)d_in[1];
  const float* embed = (const float*)d_in[2];
  const float* we_same = (const float*)d_in[3];
  const float* we_anti = (const float*)d_in[4];
  const float* we_ne = (const float*)d_in[5];
  const float* w_up = (const float*)d_in[6];
  const float* b_up = (const float*)d_in[7];
  const int* atom_types = (const int*)d_in[8];

  const int useWs = (d_ws != nullptr && ws_size >= WPK_BYTES) ? 1 : 0;
  if (useWs) {
    // 12288 (frag,lane) threads
    pack_wup_kernel<<<48, 256, 0, stream>>>(w_up, (unsigned short*)d_ws);
  }
  egnn_kernel<<<NB, 256, 0, stream>>>(r, R, embed, we_same, we_anti, we_ne,
                                      w_up, b_up, atom_types,
                                      (const unsigned short*)d_ws, useWs,
                                      (float*)d_out);
}
